// SelfAttention_1571958030928
// MI455X (gfx1250) — matrix operand order repaired and checked by reading, not compiled, not run on hardware
//
#include <hip/hip_runtime.h>
#include <hip/hip_bf16.h>

// ---------------------------------------------------------------------------
// Self-attention (B=16, C=256, CK=32, N=4096), MI455X / gfx1250.
// Compute-bound (~165 GFLOP vs ~128 MiB HBM) -> bf16 WMMA everywhere,
// flash-attention fusion, double-buffered async K/V tiles (64 keys) in LDS.
//
// Workspace (~40.2 MiB):
//   qbuf: [16][4096][32]  bf16   kbuf: [16][4096][32] bf16
//   vbuf: [16][256][4096] bf16   packed bf16 weights: Wq/Wk (16KB ea), Wv (128KB)
// ---------------------------------------------------------------------------

typedef __attribute__((ext_vector_type(16))) __bf16 v16bf;
typedef __attribute__((ext_vector_type(8)))  __bf16 v8bf;
typedef __attribute__((ext_vector_type(4)))  __bf16 v4bf;
typedef __attribute__((ext_vector_type(8)))  float  v8f;

#define WMMA_BF16(A, B, C) \
    __builtin_amdgcn_wmma_f32_16x16x32_bf16(false, (A), false, (B), (short)0, (C), false, false)

// ---- CDNA5 async global->LDS copy (ASYNCcnt-tracked DMA, §15.18.3 op 98) ----
static __device__ __forceinline__ void async_load_b128(const void* gptr, unsigned lds_byte_off) {
    asm volatile("global_load_async_to_lds_b128 %0, %1, off"
                 :: "v"(lds_byte_off), "v"(gptr)
                 : "memory");
}
static __device__ __forceinline__ void wait_asynccnt0() {
#if __has_builtin(__builtin_amdgcn_s_wait_asynccnt)
    __builtin_amdgcn_s_wait_asynccnt(0);
#else
    asm volatile("s_wait_asynccnt 0x0" ::: "memory");
#endif
}
// Generic LDS pointer -> LDS byte offset (ISA: flat LDS addr = low 32 bits).
static __device__ __forceinline__ unsigned lds_off(const void* p) {
    return (unsigned)(size_t)p;
}

// A-matrix (16x32 bf16) fragment from a contiguous 32-half row.
// Layout: lane m = lane%16, group g = lane>>4, element e -> K = 16g + e.
static __device__ __forceinline__ v16bf load_a_row32(const __bf16* row, int g) {
    const v8bf lo = *(const v8bf*)(row + 16 * g);
    const v8bf hi = *(const v8bf*)(row + 16 * g + 8);
    v16bf a;
#pragma unroll
    for (int i = 0; i < 8; ++i) { a[i] = lo[i]; a[i + 8] = hi[i]; }
    return a;
}

// ---------------------------------------------------------------------------
// Kernel 0: fp32 -> bf16 weight packing (run once, tiny).
// ---------------------------------------------------------------------------
__global__ __launch_bounds__(256)
void pack_w_kernel(const float* __restrict__ src, __bf16* __restrict__ dst, int n4) {
    const int i = blockIdx.x * 256 + threadIdx.x;
    if (i < n4) {
        const float4 f = ((const float4*)src)[i];
        v4bf o;
        o[0] = (__bf16)f.x; o[1] = (__bf16)f.y; o[2] = (__bf16)f.z; o[3] = (__bf16)f.w;
        ((v4bf*)dst)[i] = o;
    }
}

// ---------------------------------------------------------------------------
// Kernel 1: fused q/k/v projections.  grid (64,16), block 128 (4 waves).
// B-fragments come straight from pre-packed bf16 weights (32B loads).
// ---------------------------------------------------------------------------
__global__ __launch_bounds__(128)
void proj_qkv_kernel(const float* __restrict__ x,
                     const __bf16* __restrict__ wq, const float* __restrict__ bq,
                     const __bf16* __restrict__ wk, const float* __restrict__ bk,
                     const __bf16* __restrict__ wv, const float* __restrict__ bv,
                     __bf16* __restrict__ qbuf, __bf16* __restrict__ kbuf,
                     __bf16* __restrict__ vbuf) {
    constexpr int C = 256, N = 4096;
    constexpr int XS_STRIDE = 40;                 // halves; 80B rows keep 16B align
    __shared__ __attribute__((aligned(32))) __bf16 xs[64 * XS_STRIDE];

    const int tid  = threadIdx.x;
    const int lane = tid & 31;
    const int w    = tid >> 5;
    const int g    = lane >> 4;
    const int ln   = lane & 15;
    const int b    = blockIdx.y;
    const int n0   = blockIdx.x * 64;

    v8f accq[2], acck[2], accv[16];
#pragma unroll
    for (int i = 0; i < 2; ++i) { accq[i] = {}; acck[i] = {}; }
#pragma unroll
    for (int i = 0; i < 16; ++i) accv[i] = {};

    for (int c0 = 0; c0 < C; c0 += 32) {
        __syncthreads();
#pragma unroll
        for (int j = 0; j < 4; ++j) {            // x[c0..+31][n0..+63] -> bf16 LDS^T
            const int fidx = tid + 128 * j;
            const int c    = fidx >> 4;
            const int n4   = (fidx & 15) << 2;
            const float4 xv =
                *(const float4*)(x + ((size_t)b * C + c0 + c) * N + n0 + n4);
            xs[(n4 + 0) * XS_STRIDE + c] = (__bf16)xv.x;
            xs[(n4 + 1) * XS_STRIDE + c] = (__bf16)xv.y;
            xs[(n4 + 2) * XS_STRIDE + c] = (__bf16)xv.z;
            xs[(n4 + 3) * XS_STRIDE + c] = (__bf16)xv.w;
        }
        __syncthreads();

        const v16bf a = load_a_row32(xs + (16 * w + ln) * XS_STRIDE, g);

#pragma unroll
        for (int oi = 0; oi < 2; ++oi) {
            const size_t wrow = (size_t)(oi * 16 + ln) * C + c0 + 16 * g;
            accq[oi] = WMMA_BF16(a, *(const v16bf*)(wq + wrow), accq[oi]);
            acck[oi] = WMMA_BF16(a, *(const v16bf*)(wk + wrow), acck[oi]);
        }
#pragma unroll
        for (int t = 0; t < 16; ++t) {
            const size_t wrow = (size_t)(t * 16 + ln) * C + c0 + 16 * g;
            accv[t] = WMMA_BF16(a, *(const v16bf*)(wv + wrow), accv[t]);
        }
    }

    // Epilogue.  C-fragment: VGPR r, group g -> row M = r + 8g, col = ln.
    const int nb = n0 + 16 * w;
#pragma unroll
    for (int oi = 0; oi < 2; ++oi) {
        const float biq = bq[oi * 16 + ln];
        const float bik = bk[oi * 16 + ln];
#pragma unroll
        for (int r = 0; r < 8; ++r) {
            const size_t row = (size_t)b * N + nb + r + 8 * g;
            qbuf[row * 32 + oi * 16 + ln] = (__bf16)(accq[oi][r] + biq);
            kbuf[row * 32 + oi * 16 + ln] = (__bf16)(acck[oi][r] + bik);
        }
    }
#pragma unroll
    for (int t = 0; t < 16; ++t) {
        const int   ch   = t * 16 + ln;
        const float bias = bv[ch];
        v8bf o8;
#pragma unroll
        for (int r = 0; r < 8; ++r) o8[r] = (__bf16)(accv[t][r] + bias);
        *(v8bf*)(vbuf + ((size_t)b * C + ch) * N + nb + 8 * g) = o8;   // contiguous 16B
    }
}

// ---------------------------------------------------------------------------
// Kernel 2: flash attention + alpha*out + x.  grid (32,16), block 256 (8 waves).
// Each wave: 16 queries, 16x256 fp32 O-tile in VGPRs.  64-key tiles:
// K 64x32 + V 256x64 bf16 = 36KB, double-buffered via async DMA, shared by
// all 8 waves.  Per 64 keys: 4 S-WMMAs + 32 PV-WMMAs; O-rescale/shuffle/
// barrier overhead amortized over twice as many keys as a 32-key tile.
// ---------------------------------------------------------------------------
__global__ __launch_bounds__(256)
void attn_kernel(const float* __restrict__ x, const float* __restrict__ alphap,
                 const __bf16* __restrict__ qbuf, const __bf16* __restrict__ kbuf,
                 const __bf16* __restrict__ vbuf, float* __restrict__ out) {
    constexpr int C = 256, N = 4096, TK = 64;
    constexpr int KHALF  = 64 * 32;                // K tile halves (4KB)
    constexpr int KVHALF = KHALF + 256 * 64;       // + V tile (32KB) = 36KB
    __shared__ __attribute__((aligned(32))) __bf16 kv[2][KVHALF];
    __shared__ __attribute__((aligned(32))) __bf16 pstage[8][16 * 64];

    const int tid  = threadIdx.x;
    const int lane = tid & 31;
    const int w    = tid >> 5;
    const int g    = lane >> 4;
    const int ln   = lane & 15;
    const int b    = blockIdx.y;
    const int n0   = blockIdx.x * 128 + 16 * w;    // this wave's 16 queries

    const __bf16* kg_base = kbuf + (size_t)b * N * 32;
    const __bf16* vg_base = vbuf + (size_t)b * C * N;

    // Issue one K/V tile (keys m0..m0+63) into LDS buffer `buf` via async DMA.
    auto issue_tile = [&](int m0, int buf) {
        const unsigned kofs = lds_off(&kv[buf][0]);
        const unsigned vofs = kofs + KHALF * 2;
        const __bf16* kg = kg_base + (size_t)m0 * 32;
        const __bf16* vg = vg_base + m0;
#pragma unroll
        for (int j = 0; j < 8; ++j) {              // 2048 V chunks of 16B
            const int q = tid + 256 * j;
            const int ch = q >> 3, part = q & 7;   // 8 chunks per 128B channel row
            async_load_b128(vg + (size_t)ch * N + part * 8,
                            vofs + (unsigned)(ch * 64 + part * 8) * 2);
        }
        {                                          // 256 K chunks (all threads)
            const int key = tid >> 2, part = tid & 3;
            async_load_b128(kg + key * 32 + part * 8,
                            kofs + (unsigned)(key * 32 + part * 8) * 2);
        }
    };

    // Q A-fragment, loaded once.
    const v16bf aq = load_a_row32(qbuf + ((size_t)b * N + n0 + ln) * 32, g);

    float m_i[8], l_i[8];
    v8f   O[16];
#pragma unroll
    for (int r = 0; r < 8; ++r) { m_i[r] = -3.0e38f; l_i[r] = 0.0f; }
#pragma unroll
    for (int t = 0; t < 16; ++t) O[t] = {};

    __bf16* pw = pstage[w];

    issue_tile(0, 0);
    int buf = 0;
    for (int it = 0; it < N / TK; ++it) {
        wait_asynccnt0();                          // tile `it` landed in LDS
        __syncthreads();                           // visible to all 8 waves
        if (it + 1 < N / TK) issue_tile((it + 1) * TK, buf ^ 1);  // overlap DMA

        const __bf16* klds = kv[buf];
        const __bf16* vlds = kv[buf] + KHALF;

        // ---- S = Q K^T for 64 keys (4 C-fragments) ----
        v8f s[4];
#pragma unroll
        for (int sub = 0; sub < 4; ++sub) {
            v8f z = {};
            s[sub] = WMMA_BF16(
                aq, *(const v16bf*)(klds + (size_t)(sub * 16 + ln) * 32 + 16 * g), z);
        }

        // ---- online softmax; P stored to LDS as bf16 row-by-row ----
        float sc[8];
#pragma unroll
        for (int r = 0; r < 8; ++r) {
            float mt = fmaxf(fmaxf(s[0][r], s[1][r]), fmaxf(s[2][r], s[3][r]));
#pragma unroll
            for (int off = 8; off >= 1; off >>= 1)
                mt = fmaxf(mt, __shfl_xor(mt, off, 32));
            const float mn = fmaxf(m_i[r], mt);
            sc[r]  = __expf(m_i[r] - mn);
            m_i[r] = mn;
            float p0 = __expf(s[0][r] - mn);
            float p1 = __expf(s[1][r] - mn);
            float p2 = __expf(s[2][r] - mn);
            float p3 = __expf(s[3][r] - mn);
            float rs = (p0 + p1) + (p2 + p3);
#pragma unroll
            for (int off = 8; off >= 1; off >>= 1)
                rs += __shfl_xor(rs, off, 32);
            l_i[r] = l_i[r] * sc[r] + rs;
            __bf16* prow = pw + (r + 8 * g) * 64 + ln;
            prow[0]  = (__bf16)p0;
            prow[16] = (__bf16)p1;
            prow[32] = (__bf16)p2;
            prow[48] = (__bf16)p3;
        }
#pragma unroll
        for (int t = 0; t < 16; ++t)
#pragma unroll
            for (int r = 0; r < 8; ++r) O[t][r] *= sc[r];

        // ---- C-layout -> A-layout via per-wave LDS round trip ----
        asm volatile("" ::: "memory");
        __builtin_amdgcn_wave_barrier();
        const v16bf ap0 = load_a_row32(pw + ln * 64, g);        // keys  0..31
        const v16bf ap1 = load_a_row32(pw + ln * 64 + 32, g);   // keys 32..63
        asm volatile("" ::: "memory");
        __builtin_amdgcn_wave_barrier();

        // ---- O += P * V (16 channel fragments x 2 key halves) ----
#pragma unroll
        for (int t = 0; t < 16; ++t) {
            const __bf16* vrow = vlds + (size_t)(t * 16 + ln) * 64 + 16 * g;
            O[t] = WMMA_BF16(ap0, *(const v16bf*)(vrow), O[t]);
            O[t] = WMMA_BF16(ap1, *(const v16bf*)(vrow + 32), O[t]);
        }

        __syncthreads();                           // all reads of `buf` done
        buf ^= 1;
    }

    // ---- epilogue: out = alpha * (O / l) + x ----
    const float alpha = alphap[0];
    float il[8];
#pragma unroll
    for (int r = 0; r < 8; ++r) il[r] = alpha / l_i[r];

#pragma unroll
    for (int t = 0; t < 16; ++t) {
        const int    ch   = t * 16 + ln;
        const size_t base = ((size_t)b * C + ch) * N + n0 + 8 * g;   // rows contiguous
        const float4 x0 = *(const float4*)(x + base);
        const float4 x1 = *(const float4*)(x + base + 4);
        float4 y0, y1;
        y0.x = O[t][0] * il[0] + x0.x;  y0.y = O[t][1] * il[1] + x0.y;
        y0.z = O[t][2] * il[2] + x0.z;  y0.w = O[t][3] * il[3] + x0.w;
        y1.x = O[t][4] * il[4] + x1.x;  y1.y = O[t][5] * il[5] + x1.y;
        y1.z = O[t][6] * il[6] + x1.z;  y1.w = O[t][7] * il[7] + x1.w;
        *(float4*)(out + base)     = y0;
        *(float4*)(out + base + 4) = y1;
    }
}

// ---------------------------------------------------------------------------
extern "C" void kernel_launch(void* const* d_in, const int* in_sizes, int n_in,
                              void* d_out, int out_size, void* d_ws, size_t ws_size,
                              hipStream_t stream) {
    const float* x     = (const float*)d_in[0];
    const float* Wq    = (const float*)d_in[1];
    const float* bq    = (const float*)d_in[2];
    const float* Wk    = (const float*)d_in[3];
    const float* bk    = (const float*)d_in[4];
    const float* Wv    = (const float*)d_in[5];
    const float* bv    = (const float*)d_in[6];
    const float* alpha = (const float*)d_in[7];
    float*       out   = (float*)d_out;

    const size_t B = 16, C = 256, N = 4096, CK = 32;
    __bf16* qb = (__bf16*)d_ws;                 //  4 MiB
    __bf16* kb = qb + B * N * CK;               //  4 MiB
    __bf16* vb = kb + B * N * CK;               // 32 MiB
    __bf16* wq = vb + B * C * N;                // 16 KB
    __bf16* wk = wq + CK * C;                   // 16 KB
    __bf16* wv = wk + CK * C;                   // 128 KB

    // one-shot bf16 weight packing
    pack_w_kernel<<<dim3((CK * C / 4 + 255) / 256), dim3(256), 0, stream>>>(Wq, wq, CK * C / 4);
    pack_w_kernel<<<dim3((CK * C / 4 + 255) / 256), dim3(256), 0, stream>>>(Wk, wk, CK * C / 4);
    pack_w_kernel<<<dim3((C * C / 4 + 255) / 256),  dim3(256), 0, stream>>>(Wv, wv, C * C / 4);

    proj_qkv_kernel<<<dim3(64, 16), dim3(128), 0, stream>>>(x, wq, bq, wk, bk, wv, bv, qb, kb, vb);
    attn_kernel<<<dim3(32, 16), dim3(256), 0, stream>>>(x, alpha, qb, kb, vb, out);
}
